// ModulatedConv2d_5299989643716
// MI455X (gfx1250) — matrix operand order repaired and checked by reading, not compiled, not run on hardware
//
#include <hip/hip_runtime.h>

// ---------------------------------------------------------------------------
// ModulatedConv2d (StyleGAN2) for MI455X / gfx1250
//   stage 1: style modulation s[b,i]
//   stage 2: demod[b,o]
//   stage 2.5 (if ws allows): prep bf16 weights  w*s*conv_scale*demod -> ws
//   stage 3: conv as implicit GEMM  M=512 N=4096 K=4608 per batch,
//            v_wmma_f32_16x16x32_bf16, f32 accumulation.
//            BM=256 BN=128 BK=64, 8 waves (4M x 2N), 32 WMMA/wave/K-step,
//            double-buffered LDS, async global->LDS copy for A tiles.
// ---------------------------------------------------------------------------

typedef __attribute__((ext_vector_type(16))) __bf16        v16bf;
typedef __attribute__((ext_vector_type(2)))  __bf16        v2bf;
typedef __attribute__((ext_vector_type(8)))  float         v8f;
typedef __attribute__((ext_vector_type(4)))  unsigned int  v4u;
typedef __attribute__((ext_vector_type(4)))  int           v4i;

#define BATCH   8
#define IN_CH   512
#define OUT_CH  512
#define HW      64
#define NPIX    4096            // 64*64
#define KTOT    4608            // IN_CH * 9
#define BM      256
#define BN      128
#define BK      64
#define KSTEPS  (KTOT / BK)     // 72

#define AS_STRIDE 36            // dwords per A row (32 + 4 pad, bank spread)
#define BS_STRIDE 36            // dwords per B row

#define LIN_SCALE   0.04419417382415922f     // 1/sqrt(512)
#define CONV_SCALE  0.014731391274719739f    // 1/sqrt(512*9)
#define EPS_DEMOD   1e-8f

#if __has_builtin(__builtin_amdgcn_global_load_async_to_lds_b128)
#define USE_ASYNC_LDS 1
#else
#define USE_ASYNC_LDS 0
#endif

__device__ __forceinline__ unsigned int pack_bf16(float f0, float f1) {
    v2bf h;
    h[0] = (__bf16)f0;
    h[1] = (__bf16)f1;
    return __builtin_bit_cast(unsigned int, h);
}

__device__ __forceinline__ void wait_async_lds() {
#if USE_ASYNC_LDS
#if __has_builtin(__builtin_amdgcn_s_wait_asynccnt)
    __builtin_amdgcn_s_wait_asynccnt(0);
#else
    asm volatile("s_wait_asynccnt 0x0" ::: "memory");
#endif
#endif
}

// A row copy: 64 bf16 (128 B) global -> LDS
__device__ __forceinline__ void copy_a_row(const unsigned short* __restrict__ src,
                                           unsigned int* dstAs) {
#if USE_ASYNC_LDS
    #pragma unroll
    for (int j = 0; j < 8; ++j)
        __builtin_amdgcn_global_load_async_to_lds_b128(
            (__attribute__((address_space(1))) v4i*)(src + j * 8),
            (__attribute__((address_space(3))) v4i*)(dstAs + j * 4),
            0, 0);
#else
    #pragma unroll
    for (int j = 0; j < 8; ++j)
        *(v4u*)(dstAs + j * 4) = *(const v4u*)(src + j * 8);
#endif
}

// on-the-fly modulated A row (fallback when ws too small for prepped weights)
__device__ __forceinline__ void mod_a_row(const float* __restrict__ wp,
                                          const float* __restrict__ sb,
                                          int k0, unsigned int* dstAs) {
    unsigned int ich = (unsigned int)k0 / 9u;
    unsigned int rem = (unsigned int)k0 - ich * 9u;
    float sv = sb[ich];
    unsigned int pk[32];
    #pragma unroll
    for (int e = 0; e < 32; ++e) {
        float f0 = wp[2 * e] * sv;
        if (++rem == 9u) { rem = 0u; sv = sb[++ich]; }
        float f1 = wp[2 * e + 1] * sv;
        if (++rem == 9u) { rem = 0u; sv = sb[++ich]; }
        pk[e] = pack_bf16(f0, f1);
    }
    #pragma unroll
    for (int j = 0; j < 8; ++j)
        *(v4u*)(dstAs + j * 4) = *(const v4u*)&pk[j * 4];
}

// im2col gather: 32 consecutive K values for one pixel column, zero-padded
__device__ __forceinline__ void im2col_load(const float* __restrict__ xb,
                                            int yy, int xx, int kstart,
                                            float* __restrict__ breg) {
    unsigned int ich = (unsigned int)kstart / 9u;
    unsigned int rem = (unsigned int)kstart - ich * 9u;
    int t3 = (int)(rem / 3u);                      // ky
    int t1 = (int)(rem - (unsigned int)t3 * 3u);   // kx
    #pragma unroll
    for (int e = 0; e < 32; ++e) {
        int iy = yy + t3 - 1, ix = xx + t1 - 1;
        breg[e] = ((unsigned)iy < (unsigned)HW && (unsigned)ix < (unsigned)HW)
                  ? xb[(size_t)ich * NPIX + iy * HW + ix] : 0.f;
        if (++t1 == 3) { t1 = 0; if (++t3 == 3) { t3 = 0; ++ich; } }
    }
}

__device__ __forceinline__ void pack_store_b(const float* __restrict__ breg,
                                             unsigned int* BsBuf,
                                             int nB, int khB) {
    unsigned int pk[16];
    #pragma unroll
    for (int e = 0; e < 16; ++e)
        pk[e] = pack_bf16(breg[2 * e], breg[2 * e + 1]);
    #pragma unroll
    for (int j = 0; j < 4; ++j)
        *(v4u*)&BsBuf[nB * BS_STRIDE + (khB >> 1) + j * 4] =
            *(const v4u*)&pk[j * 4];
}

// --------------------------- stage 1: modulation ---------------------------
__global__ __launch_bounds__(256)
void style_mod_kernel(const float* __restrict__ style,
                      const float* __restrict__ mod_w,
                      const float* __restrict__ mod_b,
                      float* __restrict__ s_out) {
    int idx = blockIdx.x * blockDim.x + threadIdx.x;   // b*512 + i
    if (idx >= BATCH * IN_CH) return;
    int b = idx >> 9;
    int i = idx & 511;
    const float* st = style + b * 512;
    const float* mw = mod_w + (size_t)i * 512;
    float acc = 0.f;
    #pragma unroll 8
    for (int j = 0; j < 512; ++j) acc += st[j] * mw[j];
    s_out[idx] = acc * LIN_SCALE + mod_b[i];
}

// --------------------------- stage 2: demodulation -------------------------
__global__ __launch_bounds__(256)
void demod_kernel(const float* __restrict__ weight,
                  const float* __restrict__ s_in,
                  float* __restrict__ demod) {
    const int o = blockIdx.x;
    const int b = blockIdx.y;
    const float* wp = weight + (size_t)o * KTOT;
    const float* sb = s_in + b * IN_CH;
    float sum = 0.f;
    for (int k = threadIdx.x; k < KTOT; k += 256) {
        float w = wp[k] * sb[k / 9];
        sum += w * w;
    }
    #pragma unroll
    for (int off = 16; off > 0; off >>= 1)
        sum += __shfl_xor(sum, off, 32);
    __shared__ float red[8];
    if ((threadIdx.x & 31) == 0) red[threadIdx.x >> 5] = sum;
    __syncthreads();
    if (threadIdx.x == 0) {
        float t = 0.f;
        #pragma unroll
        for (int i = 0; i < 8; ++i) t += red[i];
        demod[b * OUT_CH + o] =
            rsqrtf(CONV_SCALE * CONV_SCALE * t + EPS_DEMOD);
    }
}

// ------------------- stage 2.5: prepped bf16 weights -----------------------
__global__ __launch_bounds__(256)
void prep_weight_kernel(const float* __restrict__ weight,
                        const float* __restrict__ s_in,
                        const float* __restrict__ demod,
                        unsigned short* __restrict__ wprep) {
    const int o = blockIdx.x;
    const int b = blockIdx.y;
    const float* wp = weight + (size_t)o * KTOT;
    const float* sb = s_in + b * IN_CH;
    const float dscale = CONV_SCALE * demod[b * OUT_CH + o];
    unsigned short* dst = wprep + ((size_t)b * OUT_CH + o) * KTOT;
    for (int k = threadIdx.x; k < KTOT; k += 256) {
        float f = wp[k] * sb[k / 9] * dscale;
        dst[k] = (unsigned short)(pack_bf16(f, f) & 0xFFFFu);
    }
}

// --------------------------- stage 3: conv as GEMM -------------------------
template <bool PREPPED>
__global__ __launch_bounds__(256)
void modconv_gemm_kernel(const float* __restrict__ x,
                         const float* __restrict__ weight,
                         const unsigned short* __restrict__ wprep,
                         const float* __restrict__ s_in,
                         const float* __restrict__ demod,
                         float* __restrict__ out) {
    // double-buffered bf16 tiles (108 KB total of 320 KB/WGP)
    __shared__ __align__(16) unsigned int As[2][BM * AS_STRIDE];  // 2x36 KB
    __shared__ __align__(16) unsigned int Bs[2][BN * BS_STRIDE];  // 2x18 KB

    const int b     = blockIdx.z;
    const int oBase = blockIdx.y * BM;
    const int p0    = blockIdx.x * BN;

    const int tid  = threadIdx.x;
    const int lane = tid & 31;
    const int wv   = tid >> 5;        // 8 waves
    const int wv_m = wv & 3;          // 4 waves along M (64 rows each)
    const int wv_n = wv >> 2;         // 2 waves along N (64 cols each)

    const float* xb = x + (size_t)b * IN_CH * NPIX;
    const float* sb = s_in + b * IN_CH;

    const unsigned short* aSrc =
        wprep + ((size_t)b * OUT_CH + (oBase + tid)) * KTOT;   // PREPPED
    const float* wRow = weight + (size_t)(oBase + tid) * KTOT; // fallback

    v8f acc[4][4];
    #pragma unroll
    for (int mi = 0; mi < 4; ++mi)
        #pragma unroll
        for (int ni = 0; ni < 4; ++ni)
            acc[mi][ni] = (v8f)(0.0f);

    // B loader: 2 threads per pixel column, 32 consecutive K each
    const int nB  = tid >> 1;              // 0..127
    const int khB = (tid & 1) * 32;        // 0 or 32
    const int yy  = (p0 + nB) >> 6;
    const int xx  = (p0 + nB) & 63;

    float breg[32];

    // ---- prologue: stage K-step 0 into buffer 0 ----
    if constexpr (PREPPED) copy_a_row(aSrc, &As[0][tid * AS_STRIDE]);
    else                   mod_a_row(wRow, sb, 0, &As[0][tid * AS_STRIDE]);
    im2col_load(xb, yy, xx, khB, breg);
    pack_store_b(breg, Bs[0], nB, khB);
    wait_async_lds();
    __syncthreads();

    for (int kc = 0; kc < KSTEPS; ++kc) {
        const int cur  = kc & 1;
        const int nxt  = cur ^ 1;
        const bool more = (kc + 1 < KSTEPS);
        const int k1 = (kc + 1) * BK;

        // ---- issue next tile's loads before computing current ----
        if (more) {
            if constexpr (PREPPED) {
                copy_a_row(aSrc + k1, &As[nxt][tid * AS_STRIDE]);     // async
            } else {
                __builtin_prefetch(wRow + k1, 0, 0);
            }
            im2col_load(xb, yy, xx, k1 + khB, breg);                  // to regs
        }

        // ---- 32 WMMAs on current buffer ----
        {
            const unsigned int* Ac = As[cur];
            const unsigned int* Bc = Bs[cur];
            union FragU { v4u u[2]; v16bf v; };
            const int mlane = lane & 15;
            const int aOff  = (lane >> 4) * 4;   // A: K base 0 / 8  (dwords)
            const int bOff  = (lane >> 4) * 4;   // B: K base 0 / 8  (dwords)

            #pragma unroll
            for (int c = 0; c < 2; ++c) {        // two K=32 chunks per BK=64
                const int cu = c * 16;
                FragU fa[4], fb[4];
                #pragma unroll
                for (int mi = 0; mi < 4; ++mi) {
                    int mrow = wv_m * 64 + mi * 16 + mlane;
                    fa[mi].u[0] = *(const v4u*)&Ac[mrow * AS_STRIDE + cu + aOff];
                    fa[mi].u[1] = *(const v4u*)&Ac[mrow * AS_STRIDE + cu + aOff + 8];
                }
                #pragma unroll
                for (int ni = 0; ni < 4; ++ni) {
                    int nrow = wv_n * 64 + ni * 16 + mlane;
                    fb[ni].u[0] = *(const v4u*)&Bc[nrow * BS_STRIDE + cu + bOff];
                    fb[ni].u[1] = *(const v4u*)&Bc[nrow * BS_STRIDE + cu + bOff + 8];
                }
                #pragma unroll
                for (int mi = 0; mi < 4; ++mi)
                    #pragma unroll
                    for (int ni = 0; ni < 4; ++ni)
                        acc[mi][ni] = __builtin_amdgcn_wmma_f32_16x16x32_bf16(
                            false, fa[mi].v, false, fb[ni].v,
                            (short)0, acc[mi][ni], false, false);
            }
        }

        // ---- finish staging next buffer ----
        if (more) {
            if constexpr (!PREPPED)
                mod_a_row(wRow + k1, sb, k1, &As[nxt][tid * AS_STRIDE]);
            pack_store_b(breg, Bs[nxt], nB, khB);
        }
        wait_async_lds();
        __syncthreads();
    }

    // ---- epilogue ----
    const float* dm = demod + b * OUT_CH;
    float* ob = out + (size_t)b * OUT_CH * NPIX + p0;
    #pragma unroll
    for (int mi = 0; mi < 4; ++mi) {
        int oSub = oBase + wv_m * 64 + mi * 16 + ((lane >> 4) << 3);
        #pragma unroll
        for (int ni = 0; ni < 4; ++ni) {
            int nn = wv_n * 64 + ni * 16 + (lane & 15);
            #pragma unroll
            for (int vg = 0; vg < 8; ++vg) {
                int o = oSub + vg;     // C layout: M = vg + 8*(lane>=16)
                float v = acc[mi][ni][vg];
                if constexpr (!PREPPED) v *= CONV_SCALE * dm[o];
                ob[(size_t)o * NPIX + nn] = v;
            }
        }
    }
}

// ---------------------------------------------------------------------------
extern "C" void kernel_launch(void* const* d_in, const int* in_sizes, int n_in,
                              void* d_out, int out_size, void* d_ws, size_t ws_size,
                              hipStream_t stream) {
    const float* x      = (const float*)d_in[0];  // [8,512,64,64]
    const float* style  = (const float*)d_in[1];  // [8,512]
    const float* weight = (const float*)d_in[2];  // [1,512,512,3,3]
    const float* mod_w  = (const float*)d_in[3];  // [512,512]
    const float* mod_b  = (const float*)d_in[4];  // [512]
    float* out = (float*)d_out;                   // [8,512,64,64]

    float* wsf       = (float*)d_ws;
    float* s_buf     = wsf;                        // 8*512 floats
    float* demod_buf = wsf + BATCH * IN_CH;        // 8*512 floats
    unsigned short* wprep =
        (unsigned short*)((char*)d_ws + 32768);    // prepped bf16 weights

    const size_t need = 32768 + (size_t)BATCH * OUT_CH * KTOT * sizeof(unsigned short);

    style_mod_kernel<<<(BATCH * IN_CH + 255) / 256, 256, 0, stream>>>(
        style, mod_w, mod_b, s_buf);

    demod_kernel<<<dim3(OUT_CH, BATCH), 256, 0, stream>>>(
        weight, s_buf, demod_buf);

    dim3 grid(NPIX / BN, OUT_CH / BM, BATCH);
    if (ws_size >= need) {
        prep_weight_kernel<<<dim3(OUT_CH, BATCH), 256, 0, stream>>>(
            weight, s_buf, demod_buf, wprep);
        modconv_gemm_kernel<true><<<grid, 256, 0, stream>>>(
            x, weight, wprep, s_buf, demod_buf, out);
    } else {
        modconv_gemm_kernel<false><<<grid, 256, 0, stream>>>(
            x, weight, wprep, s_buf, demod_buf, out);
    }
}
